// GraphConv0TPK_79250736546092
// MI455X (gfx1250) — matrix operand order repaired and checked by reading, compiled, not run on hardware
//
#include <hip/hip_runtime.h>
#include <hip/hip_bf16.h>
#include <math.h>

// Problem constants (match reference)
#define NN 100000
#define EE 1600000
#define FF 128
#define HH 128
#define GG 64
#define CC 10

typedef __attribute__((ext_vector_type(16))) __bf16 v16bf;
typedef __attribute__((ext_vector_type(8)))  float  v8f;

// LDS tile row stride in halves: 136 -> 272 B/row = 17*16 (B128-aligned rows,
// 68 dwords/row rotates banks by 4 per row => conflict-free A-fragment reads)
#define TSTR 136

__device__ __forceinline__ unsigned short f2bf(float f) {
    union { float f; unsigned int u; } c; c.f = f;
    unsigned int u = c.u;
    unsigned int r = u + 0x7FFFu + ((u >> 16) & 1u);   // round-to-nearest-even
    return (unsigned short)(r >> 16);
}

__device__ __forceinline__ float bf2f(unsigned short h) {
    union { float f; unsigned int u; } c;
    c.u = ((unsigned int)h) << 16;
    return c.f;
}

// ---------------------------------------------------------------------------
// Zero a float buffer
// ---------------------------------------------------------------------------
__global__ void zero_f32(float* __restrict__ p, int n) {
    int i = blockIdx.x * blockDim.x + threadIdx.x;
    if (i < n) p[i] = 0.0f;
}

// ---------------------------------------------------------------------------
// fp32 -> bf16 conversion (x -> h0)
// ---------------------------------------------------------------------------
__global__ void f32_to_bf16(const float* __restrict__ in,
                            unsigned short* __restrict__ out, int n) {
    int i = blockIdx.x * blockDim.x + threadIdx.x;
    if (i < n) out[i] = f2bf(in[i]);
}

// ---------------------------------------------------------------------------
// Weight prep: fp32 [128 out][128 in] -> bf16 in WMMA B-operand layout.
// Layout per matrix: [kt(4)][nt(8)][lane(32)][half(16)]
//   n = nt*16 + (lane & 15)
//   k = kt*32 + half + 16*(lane >> 4)
//   value = W[n][k]   (B[k][n] = W[n][k] since out = A @ W^T)
// ---------------------------------------------------------------------------
struct WPtrs { const float* w[6]; };

__global__ void prep_weights(WPtrs wp, unsigned short* __restrict__ Wbf) {
    int idx = blockIdx.x * blockDim.x + threadIdx.x;
    if (idx >= 6 * 16384) return;
    int mat  = idx / 16384;
    int j    = idx & 16383;
    int kt   = j >> 12;
    int r    = j & 4095;
    int nt   = r >> 9;
    int q    = r & 511;
    int lane = q >> 4;
    int half = q & 15;
    int n = nt * 16 + (lane & 15);
    int k = kt * 32 + half + 16 * (lane >> 4);
    Wbf[idx] = f2bf(wp.w[mat][n * 128 + k]);
}

// ---------------------------------------------------------------------------
// Edge aggregation: one wave per edge; 32 lanes x 4 bf16 = 128 features.
// agg[dst[e]][:] += h[src[e]][:]   (fp32 accumulation via global atomics)
// ---------------------------------------------------------------------------
__global__ __launch_bounds__(256) void aggregate(
    const unsigned short* __restrict__ h, const int* __restrict__ src,
    const int* __restrict__ dst, float* __restrict__ agg)
{
    int gid  = blockIdx.x * blockDim.x + threadIdx.x;
    int e    = gid >> 5;
    int lane = gid & 31;
    if (e >= EE) return;
    int s = __builtin_amdgcn_readfirstlane(src[e]);   // wave-uniform -> SGPR
    int d = __builtin_amdgcn_readfirstlane(dst[e]);
    // 8 B = 4 bf16 per lane, fully coalesced 256 B/wave
    const uint2 v = ((const uint2*)(h + (size_t)s * FF))[lane];
    float* o = agg + (size_t)d * FF + lane * 4;
    unsafeAtomicAdd(o + 0, bf2f((unsigned short)(v.x & 0xFFFFu)));
    unsafeAtomicAdd(o + 1, bf2f((unsigned short)(v.x >> 16)));
    unsafeAtomicAdd(o + 2, bf2f((unsigned short)(v.y & 0xFFFFu)));
    unsafeAtomicAdd(o + 3, bf2f((unsigned short)(v.y >> 16)));
}

// ---------------------------------------------------------------------------
// Fused GEMM: hnext = bf16(relu(agg @ Wrel^T + hcur @ Wroot^T + bias))
// Block = 256 threads (8 waves). Block owns 16 node rows; wave w owns
// output columns [w*16, w*16+16). K=128 in 4 WMMA steps per matrix.
// h-tile staged with CDNA5 async global->LDS copy (ASYNCcnt); agg tile
// converted fp32->bf16 through VGPRs concurrently.
// ---------------------------------------------------------------------------
union AFrag { v16bf v; unsigned int u[8]; };

__global__ __launch_bounds__(256) void gemm_bias_relu(
    const unsigned short* __restrict__ hcur, const float* __restrict__ agg,
    const unsigned short* __restrict__ Brel,
    const unsigned short* __restrict__ Broot,
    const float* __restrict__ bias, unsigned short* __restrict__ hnext)
{
    __shared__ unsigned short sA[16 * TSTR];
    __shared__ unsigned short sH[16 * TSTR];

    const int tid     = threadIdx.x;
    const int rowbase = blockIdx.x * 16;

    // --- h tile: async copy (already bf16 in global). 256 threads x 16 B.
    {
        int row   = tid >> 4;     // 0..15
        int chunk = tid & 15;     // 16 B chunks within a 256 B row
        unsigned long long ga = (unsigned long long)(const void*)
            (hcur + (size_t)(rowbase + row) * HH + chunk * 8);
        unsigned lda = (unsigned)(size_t)(&sH[row * TSTR + chunk * 8]);
        asm volatile("global_load_async_to_lds_b128 %0, %1, off"
                     :: "v"(lda), "v"(ga) : "memory");
    }

    // --- agg tile: load fp32, round to bf16, store to LDS
#pragma unroll
    for (int i = 0; i < 8; ++i) {
        int idx = tid + 256 * i;
        int r = idx >> 7, c = idx & 127;
        sA[r * TSTR + c] = f2bf(agg[(size_t)(rowbase + r) * FF + c]);
    }

    asm volatile("s_wait_asynccnt 0" ::: "memory");
    __syncthreads();

    const int nt   = tid >> 5;       // wave id = output column tile
    const int lane = tid & 31;
    const int nloc = lane & 15;      // A-row (M) and D-column (N)
    const int grp  = lane >> 4;

    v8f acc = {};

#pragma unroll
    for (int kt = 0; kt < 4; ++kt) {
        AFrag a, hfr, br, bo;
#pragma unroll
        for (int v = 0; v < 8; ++v) {
            // A layout: vgpr v holds K pair (kb, kb+1)
            int kb = 2 * v + 16 * grp + kt * 32;
            a.u[v]   = *(const unsigned int*)&sA[nloc * TSTR + kb];
            hfr.u[v] = *(const unsigned int*)&sH[nloc * TSTR + kb];
        }
        const unsigned int* pr = (const unsigned int*)(Brel  + (kt * 8 + nt) * 512 + lane * 16);
        const unsigned int* po = (const unsigned int*)(Broot + (kt * 8 + nt) * 512 + lane * 16);
#pragma unroll
        for (int v = 0; v < 8; ++v) { br.u[v] = pr[v]; bo.u[v] = po[v]; }

        acc = __builtin_amdgcn_wmma_f32_16x16x32_bf16(
            false, a.v,   false, br.v, (short)0, acc, false, false);
        acc = __builtin_amdgcn_wmma_f32_16x16x32_bf16(
            false, hfr.v, false, bo.v, (short)0, acc, false, false);
    }

    const float b = bias[nt * 16 + nloc];
#pragma unroll
    for (int r = 0; r < 8; ++r) {
        float o = acc[r] + b;
        o = o > 0.0f ? o : 0.0f;
        // D layout: lane -> N = lane&15, vgpr r -> M = r + 8*(lane>>4)
        hnext[(size_t)(rowbase + r + 8 * grp) * HH + nt * 16 + nloc] = f2bf(o);
    }
}

// ---------------------------------------------------------------------------
// Pool accumulate: gsum[batch[n]][f] += h[n][f]; gcnt[batch[n]] += 1 (f==0)
// ---------------------------------------------------------------------------
__global__ __launch_bounds__(256) void pool_accum(
    const unsigned short* __restrict__ h, const int* __restrict__ batch,
    float* __restrict__ gsum, float* __restrict__ gcnt)
{
    int idx = blockIdx.x * blockDim.x + threadIdx.x;
    if (idx >= NN * FF) return;
    int n = idx >> 7;
    int f = idx & 127;
    int b = batch[n];
    unsafeAtomicAdd(&gsum[b * FF + f], bf2f(h[idx]));
    if (f == 0) unsafeAtomicAdd(&gcnt[b], 1.0f);
}

// ---------------------------------------------------------------------------
// Head: pooled = gsum/max(cnt,1); h = relu(pooled@Wl1^T+bl1);
//       out = log_softmax(h@Wl2^T + bl2).  One block (128 threads)/graph.
// ---------------------------------------------------------------------------
__global__ __launch_bounds__(128) void head_kernel(
    const float* __restrict__ gsum, const float* __restrict__ gcnt,
    const float* __restrict__ Wl1, const float* __restrict__ bl1,
    const float* __restrict__ Wl2, const float* __restrict__ bl2,
    float* __restrict__ out)
{
    __shared__ float pooled[128];
    __shared__ float hid[64];
    __shared__ float logit[CC];
    __shared__ float red[2];

    int g = blockIdx.x;
    int t = threadIdx.x;

    float cnt = fmaxf(gcnt[g], 1.0f);
    pooled[t] = gsum[g * FF + t] / cnt;
    __syncthreads();

    if (t < 64) {
        float a = bl1[t];
        const float* w = Wl1 + t * 128;
        for (int k = 0; k < 128; ++k) a += pooled[k] * w[k];
        hid[t] = fmaxf(a, 0.0f);
    }
    __syncthreads();

    if (t < CC) {
        float a = bl2[t];
        const float* w = Wl2 + t * 64;
        for (int k = 0; k < 64; ++k) a += hid[k] * w[k];
        logit[t] = a;
    }
    __syncthreads();

    if (t == 0) {
        float m = logit[0];
        for (int i = 1; i < CC; ++i) m = fmaxf(m, logit[i]);
        float s = 0.0f;
        for (int i = 0; i < CC; ++i) s += expf(logit[i] - m);
        red[0] = m;
        red[1] = logf(s);
    }
    __syncthreads();

    if (t < CC) out[g * CC + t] = logit[t] - red[0] - red[1];
}

// ---------------------------------------------------------------------------
// Launch
// ---------------------------------------------------------------------------
extern "C" void kernel_launch(void* const* d_in, const int* in_sizes, int n_in,
                              void* d_out, int out_size, void* d_ws, size_t ws_size,
                              hipStream_t stream)
{
    (void)in_sizes; (void)n_in; (void)out_size; (void)ws_size;

    const float* x     = (const float*)d_in[0];
    const int*   ei    = (const int*)d_in[1];
    const int*   batch = (const int*)d_in[2];
    // d_in[3] = target_size scalar (unused; G is compile-time)
    const float* W1r = (const float*)d_in[4];
    const float* b1  = (const float*)d_in[5];
    const float* W1o = (const float*)d_in[6];
    const float* W2r = (const float*)d_in[7];
    const float* b2  = (const float*)d_in[8];
    const float* W2o = (const float*)d_in[9];
    const float* W3r = (const float*)d_in[10];
    const float* b3  = (const float*)d_in[11];
    const float* W3o = (const float*)d_in[12];
    const float* Wl1 = (const float*)d_in[13];
    const float* bl1 = (const float*)d_in[14];
    const float* Wl2 = (const float*)d_in[15];
    const float* bl2 = (const float*)d_in[16];

    // Workspace partition (~128.2 MB total)
    char* ws = (char*)d_ws;
    float*          aggb = (float*)(ws + 0);                        // 51,200,000 B
    unsigned short* hb0  = (unsigned short*)(ws + 51200000);        // 25,600,000 B
    unsigned short* hb1  = (unsigned short*)(ws + 76800000);        // 25,600,000 B
    unsigned short* hb2  = (unsigned short*)(ws + 102400000);       // 25,600,000 B
    unsigned short* Wbf  = (unsigned short*)(ws + 128000000);       //    196,608 B
    float*          gsum = (float*)(ws + 128196608);                //     32,768 B
    float*          gcnt = (float*)(ws + 128229376);                //        256 B

    // 1) bf16 weight prep in WMMA B layout + x -> bf16
    WPtrs wp;
    wp.w[0] = W1r; wp.w[1] = W1o;
    wp.w[2] = W2r; wp.w[3] = W2o;
    wp.w[4] = W3r; wp.w[5] = W3o;
    prep_weights<<<(6 * 16384 + 255) / 256, 256, 0, stream>>>(wp, Wbf);
    f32_to_bf16<<<(NN * FF) / 256, 256, 0, stream>>>(x, hb0, NN * FF);

    const int* src = ei;
    const int* dst = ei + EE;

    const float*    biases[3] = { b1, b2, b3 };
    const unsigned short* hins[3]  = { hb0, hb1, hb2 };
    unsigned short*       houts[3] = { hb1, hb2, hb1 };

    // 2) three GraphConv layers (fp32 atomic aggregation, WMMA bf16 GEMM)
    for (int l = 0; l < 3; ++l) {
        zero_f32<<<(NN * FF) / 256, 256, 0, stream>>>(aggb, NN * FF);
        aggregate<<<(EE * 32) / 256, 256, 0, stream>>>(hins[l], src, dst, aggb);
        gemm_bias_relu<<<NN / 16, 256, 0, stream>>>(
            hins[l], aggb,
            Wbf + (2 * l) * 16384, Wbf + (2 * l + 1) * 16384,
            biases[l], houts[l]);
    }

    // 3) mean pool (gsum and gcnt contiguous: zero both at once)
    zero_f32<<<(GG * FF + GG + 255) / 256, 256, 0, stream>>>(gsum, GG * FF + GG);
    pool_accum<<<(NN * FF) / 256, 256, 0, stream>>>(hb1, batch, gsum, gcnt);

    // 4) MLP head + log_softmax
    head_kernel<<<GG, 128, 0, stream>>>(gsum, gcnt, Wl1, bl1, Wl2, bl2,
                                        (float*)d_out);
}
